// SelfAttention_68719477385
// MI455X (gfx1250) — matrix operand order repaired and checked by reading, not compiled, not run on hardware
//
#include <hip/hip_runtime.h>
#include <hip/hip_bf16.h>
#include <math.h>

// Self-attention: B=16, L=2048, H=512, P=64 (fp32 in/out).
//   cvt_kernel  : Wo (f32) -> Woh (f16) workspace copy (TDM can't convert)
//   qkv_kernel  : [BL,512] x [512,64] x3  -> Qh,Kh,Vh (f16)    [WMMA]
//   attn_kernel : flash attention, TDM double-buffered K/V LDS  [WMMA + TDM]
//   out_kernel  : [BL,64] x [64,512] + bo -> out (f32)          [WMMA + TDM]
// attention_mask adds a uniform per-row shift before softmax (broadcasts over
// the key axis in the reference) -> softmax-invariant -> unused.
// Workspace: 4 x (32768*64) f16 + 32768 f16 = ~16.06 MB.

#define B_DIM 16
#define L_DIM 2048
#define H_DIM 512
#define P_DIM 64
#define BL (B_DIM * L_DIM)

typedef __attribute__((ext_vector_type(16))) _Float16 v16h;
typedef __attribute__((ext_vector_type(8)))  float    v8f;
typedef __attribute__((ext_vector_type(4)))  unsigned int u32x4;
typedef __attribute__((ext_vector_type(8)))  int          i32x8;
typedef __attribute__((ext_vector_type(4)))  int          i32x4;

__device__ __forceinline__ v8f wmma32(v16h a, v16h b, v8f c) {
  // D = A(16x32 f16) * B(32x16 f16) + C(16x16 f32)
  return __builtin_amdgcn_wmma_f32_16x16x32_f16(
      /*neg_a=*/false, a, /*neg_b=*/false, b,
      /*c_mod=*/(short)0, c, /*reuse_a=*/false, /*reuse_b=*/false);
}

// ---------------------------------------------------------------------------
// Tensor Data Mover: async 2D f16 tile Global -> LDS (tracked by TENSORcnt).
// D# layout per CDNA5 ISA ch.8 (group0: count/lds/global/type, group1:
// data_size + tensor/tile dims + dim0 stride).  2D tile -> groups 2/3 zero.
// ---------------------------------------------------------------------------
__device__ __forceinline__ void tdm_load_2d_f16(const _Float16* gsrc,
                                                void* lds_dst, int tile_cols,
                                                int tile_rows, int gstride,
                                                int tensor_cols,
                                                int tensor_rows) {
  const unsigned long long ga = (unsigned long long)(size_t)gsrc;
  u32x4 g0 = {};
  g0[0] = 1u;                                   // count=1 (valid user D#)
  g0[1] = (unsigned int)(size_t)lds_dst;        // lds_addr (low 32 = LDS off)
  g0[2] = (unsigned int)ga;                     // global_addr[31:0]
  g0[3] = ((unsigned int)(ga >> 32) & 0x01FFFFFFu) | (2u << 30);  // type=2
  i32x8 g1 = {};
  g1[0] = 1 << 16;                              // data_size=1 (2 bytes)
  g1[1] = (tensor_cols & 0xFFFF) << 16;                           // dim0 lo
  g1[2] = ((tensor_rows & 0xFFFF) << 16) | ((tensor_cols >> 16) & 0xFFFF);
  g1[3] = ((tile_cols & 0xFFFF) << 16) | ((tensor_rows >> 16) & 0xFFFF);
  g1[4] = tile_rows & 0xFFFF;                   // tile_dim1 (tile_dim2=0)
  g1[5] = gstride;                              // tensor_dim0_stride[31:0]
  const i32x4 z4 = {};
#if defined(__clang_major__) && (__clang_major__ >= 23)
  const i32x8 z8 = {};
  __builtin_amdgcn_tensor_load_to_lds(g0, g1, z4, z4, z8, 0);
#else
  __builtin_amdgcn_tensor_load_to_lds(g0, g1, z4, z4, 0);
#endif
}

// A fragment (16x32 f16), row-major source base[m*stride + k].
// Lanes 0-15: M=lane, K {0..15}; lanes 16-31: M=lane-16, K {16..31}.
__device__ __forceinline__ v16h load_a_frag(const _Float16* base, int stride,
                                            int m0, int k0, int lane) {
  v16h a;
  const int m  = m0 + (lane & 15);
  const int kb = k0 + ((lane & 16) ? 16 : 0);
#pragma unroll
  for (int v = 0; v < 8; ++v) {
    const int kk = kb + 2 * v;
    a[2 * v]     = base[m * stride + kk];
    a[2 * v + 1] = base[m * stride + kk + 1];
  }
  return a;
}

// B fragment (32x16 f16), row-major source B[k][n] = base[k*stride + n].
__device__ __forceinline__ v16h load_b_frag(const _Float16* base, int stride,
                                            int k0, int n0, int lane) {
  v16h b;
  const int n  = n0 + (lane & 15);
  const int kb = k0 + ((lane & 16) ? 16 : 0);
#pragma unroll
  for (int v = 0; v < 8; ++v) {
    b[2 * v]     = base[(kb + 2 * v) * stride + n];
    b[2 * v + 1] = base[(kb + 2 * v + 1) * stride + n];
  }
  return b;
}

// B fragment, transposed source: B[k][n] = base[n*stride + k]  (for Q*K^T).
__device__ __forceinline__ v16h load_bT_frag(const _Float16* base, int stride,
                                             int k0, int n0, int lane) {
  v16h b;
  const int n  = n0 + (lane & 15);
  const int kb = k0 + ((lane & 16) ? 16 : 0);
#pragma unroll
  for (int v = 0; v < 8; ++v) {
    b[2 * v]     = base[n * stride + kb + 2 * v];
    b[2 * v + 1] = base[n * stride + kb + 2 * v + 1];
  }
  return b;
}

// B fragment from an f32 row-major source, converting to f16 on the fly.
__device__ __forceinline__ v16h load_b_frag_f32(const float* base, int stride,
                                                int k0, int n0, int lane) {
  v16h b;
  const int n  = n0 + (lane & 15);
  const int kb = k0 + ((lane & 16) ? 16 : 0);
#pragma unroll
  for (int v = 0; v < 8; ++v) {
    b[2 * v]     = (_Float16)base[(kb + 2 * v) * stride + n];
    b[2 * v + 1] = (_Float16)base[(kb + 2 * v + 1) * stride + n];
  }
  return b;
}

// ---------------------------------------------------------------------------
// f32 -> f16 elementwise convert (for Wo, so TDM can stage it later).
// ---------------------------------------------------------------------------
__global__ __launch_bounds__(256) void cvt_kernel(const float* __restrict__ src,
                                                  _Float16* __restrict__ dst,
                                                  int n) {
  const int i = blockIdx.x * 256 + threadIdx.x;
  if (i < n) dst[i] = (_Float16)src[i];
}

// ---------------------------------------------------------------------------
// Kernel 1: Q/K/V projections.  32 query rows staged f32->f16 in LDS;
// 6 waves = {Q,K,V} x {row half}.  M=32/blk, N=64, K=512.
// ---------------------------------------------------------------------------
__global__ __launch_bounds__(192) void qkv_kernel(
    const float* __restrict__ query,
    const float* __restrict__ Wq, const float* __restrict__ bq,
    const float* __restrict__ Wk, const float* __restrict__ bk,
    const float* __restrict__ Wv, const float* __restrict__ bv,
    _Float16* __restrict__ Qh, _Float16* __restrict__ Kh,
    _Float16* __restrict__ Vh) {
  __shared__ __align__(128) _Float16 Qs[32 * H_DIM];  // 32 KB

  const int tid  = threadIdx.x;
  const int row0 = blockIdx.x * 32;  // global row in [0, B*L)

  for (int i = tid; i < 32 * H_DIM; i += 192) {
    Qs[i] = (_Float16)query[(size_t)row0 * H_DIM + i];
  }
  __syncthreads();

  const int wave = tid >> 5;
  const int lane = tid & 31;
  const int mat  = wave % 3;         // 0=Q 1=K 2=V
  const int m0   = (wave / 3) * 16;  // row half within tile

  const float* W    = (mat == 0) ? Wq : (mat == 1) ? Wk : Wv;
  const float* bias = (mat == 0) ? bq : (mat == 1) ? bk : bv;
  _Float16*    Dst  = (mat == 0) ? Qh : (mat == 1) ? Kh : Vh;

  v8f acc[4] = {};
  for (int k0 = 0; k0 < H_DIM; k0 += 32) {
    v16h a = load_a_frag(Qs, H_DIM, m0, k0, lane);
#pragma unroll
    for (int t = 0; t < 4; ++t) {
      v16h b = load_b_frag_f32(W, P_DIM, k0, t * 16, lane);
      acc[t] = wmma32(a, b, acc[t]);
    }
  }

  const int n_lo = lane & 15;
  const int mb   = m0 + ((lane & 16) ? 8 : 0);
#pragma unroll
  for (int t = 0; t < 4; ++t) {
    const int   n  = t * 16 + n_lo;
    const float bb = bias[n];
#pragma unroll
    for (int r = 0; r < 8; ++r) {
      const int grow = row0 + mb + r;
      Dst[(size_t)grow * P_DIM + n] = (_Float16)(acc[t][r] + bb);
    }
  }
}

// ---------------------------------------------------------------------------
// Kernel 2: flash attention.  Block = (batch, 64 query rows), 4 waves each
// owning 16 rows.  K/V 64x64 tiles DMA'd into LDS by the Tensor Data Mover,
// double-buffered (issue tile i+1, s_wait_tensorcnt<=2 -> tile i landed).
// Online softmax in registers; P goes through LDS for C->A layout conversion.
// ---------------------------------------------------------------------------
__global__ __launch_bounds__(128) void attn_kernel(
    const _Float16* __restrict__ Qh, const _Float16* __restrict__ Kh,
    const _Float16* __restrict__ Vh, _Float16* __restrict__ Wt) {
  __shared__ __align__(128) _Float16 Qs[64 * P_DIM];     // 8 KB
  __shared__ __align__(128) _Float16 Ks[2][64 * P_DIM];  // 16 KB
  __shared__ __align__(128) _Float16 Vs[2][64 * P_DIM];  // 16 KB
  __shared__ __align__(128) _Float16 Ps[4][16 * 64];     // 8 KB

  const int    b    = blockIdx.y;
  const int    q0   = blockIdx.x * 64;
  const int    tid  = threadIdx.x;
  const int    wave = tid >> 5;
  const int    lane = tid & 31;
  const size_t base = (size_t)b * L_DIM * P_DIM;

  // Prologue: TDM-stage Q tile and the first K/V tiles (wave 0 issues).
  if (wave == 0) {
    tdm_load_2d_f16(Qh + base + (size_t)q0 * P_DIM, Qs, P_DIM, 64, P_DIM,
                    P_DIM, L_DIM);
    tdm_load_2d_f16(Kh + base, &Ks[0][0], P_DIM, 64, P_DIM, P_DIM, L_DIM);
    tdm_load_2d_f16(Vh + base, &Vs[0][0], P_DIM, 64, P_DIM, P_DIM, L_DIM);
  }

  const int m0 = wave * 16;
  v8f   o[4] = {};
  float mi[8], li[8];
#pragma unroll
  for (int r = 0; r < 8; ++r) { mi[r] = -1e30f; li[r] = 0.0f; }

  for (int kt = 0; kt < L_DIM; kt += 64) {
    const int cur = (kt >> 6) & 1;
    __syncthreads();  // all waves done reading buffer 'cur' (prev round)
    if (wave == 0) {
      if (kt + 64 < L_DIM) {
        // Prefetch next K/V tiles into the other buffer, then wait until
        // only those two remain outstanding => current tiles have landed.
        tdm_load_2d_f16(Kh + base + (size_t)(kt + 64) * P_DIM,
                        &Ks[cur ^ 1][0], P_DIM, 64, P_DIM, P_DIM, L_DIM);
        tdm_load_2d_f16(Vh + base + (size_t)(kt + 64) * P_DIM,
                        &Vs[cur ^ 1][0], P_DIM, 64, P_DIM, P_DIM, L_DIM);
        __builtin_amdgcn_s_wait_tensorcnt(2);
      } else {
        __builtin_amdgcn_s_wait_tensorcnt(0);
      }
    }
    __syncthreads();  // publish LDS tiles to all waves

    // S = Q * K^T  (16x64 per wave, contraction over P)
    v8f s[4] = {};
    for (int ks = 0; ks < P_DIM; ks += 32) {
      v16h a = load_a_frag(Qs, P_DIM, m0, ks, lane);
#pragma unroll
      for (int t = 0; t < 4; ++t) {
        v16h bf = load_bT_frag(&Ks[cur][0], P_DIM, ks, t * 16, lane);
        s[t]    = wmma32(a, bf, s[t]);
      }
    }
    // 1/sqrt(P) scale
#pragma unroll
    for (int t = 0; t < 4; ++t)
#pragma unroll
      for (int r = 0; r < 8; ++r) s[t][r] *= 0.125f;

    // Online softmax: row M=r lives in lanes 0-15 (M=r+8 in lanes 16-31),
    // xor masks 1,2,4,8 keep the reduction within each 16-lane half.
#pragma unroll
    for (int r = 0; r < 8; ++r) {
      float smax = s[0][r];
#pragma unroll
      for (int t = 1; t < 4; ++t) smax = fmaxf(smax, s[t][r]);
      for (int off = 1; off < 16; off <<= 1)
        smax = fmaxf(smax, __shfl_xor(smax, off, 32));
      const float mnew  = fmaxf(mi[r], smax);
      const float scale = __expf(mi[r] - mnew);
      float rs = 0.0f;
#pragma unroll
      for (int t = 0; t < 4; ++t) {
        const float e = __expf(s[t][r] - mnew);
        s[t][r]       = e;
        rs += e;
      }
      for (int off = 1; off < 16; off <<= 1) rs += __shfl_xor(rs, off, 32);
      li[r] = li[r] * scale + rs;
      mi[r] = mnew;
#pragma unroll
      for (int t = 0; t < 4; ++t) o[t][r] *= scale;
    }

    // Stage P (C-layout -> row-major LDS) for reuse as an A fragment.
    {
      const int mb = (lane & 16) ? 8 : 0;
      const int nl = lane & 15;
#pragma unroll
      for (int t = 0; t < 4; ++t)
#pragma unroll
        for (int r = 0; r < 8; ++r)
          Ps[wave][(r + mb) * 64 + t * 16 + nl] = (_Float16)s[t][r];
    }
    __syncthreads();

    // O += P * V  (contraction over the 64 keys of this tile)
    for (int ks = 0; ks < 64; ks += 32) {
      v16h a = load_a_frag(&Ps[wave][0], 64, 0, ks, lane);
#pragma unroll
      for (int t = 0; t < 4; ++t) {
        v16h bf = load_b_frag(&Vs[cur][0], P_DIM, ks, t * 16, lane);
        o[t]    = wmma32(a, bf, o[t]);
      }
    }
  }

  // Normalize and store f16 weighted values.
  const int nl = lane & 15;
  const int mb = m0 + ((lane & 16) ? 8 : 0);
#pragma unroll
  for (int r = 0; r < 8; ++r) {
    const float inv = 1.0f / li[r];
#pragma unroll
    for (int t = 0; t < 4; ++t) {
      Wt[base + (size_t)(q0 + mb + r) * P_DIM + t * 16 + nl] =
          (_Float16)(o[t][r] * inv);
    }
  }
}

// ---------------------------------------------------------------------------
// Kernel 3: output projection.  Woh (64x512 f16) TDM-staged whole into LDS
// (64 KB, well under the 320 KB WGP budget).  Block = 32 rows; 8 waves =
// 2 row-tiles x 4 column groups of 128.
// ---------------------------------------------------------------------------
__global__ __launch_bounds__(256) void out_kernel(
    const _Float16* __restrict__ Wt, const _Float16* __restrict__ Woh,
    const float* __restrict__ bo, float* __restrict__ out) {
  __shared__ __align__(128) _Float16 WoS[P_DIM * H_DIM];  // 64 KB

  const int tid  = threadIdx.x;
  const int wave = tid >> 5;
  const int lane = tid & 31;

  if (wave == 0) {
    tdm_load_2d_f16(Woh, WoS, H_DIM, P_DIM, H_DIM, H_DIM, P_DIM);
    __builtin_amdgcn_s_wait_tensorcnt(0);
  }
  __syncthreads();

  const int row0 = blockIdx.x * 32 + (wave & 1) * 16;  // global row
  const int c0   = (wave >> 1) * 128;

  v8f acc[8] = {};
  for (int k0 = 0; k0 < P_DIM; k0 += 32) {
    v16h a = load_a_frag(Wt, P_DIM, row0, k0, lane);
#pragma unroll
    for (int t = 0; t < 8; ++t) {
      v16h b = load_b_frag(WoS, H_DIM, k0, c0 + t * 16, lane);
      acc[t] = wmma32(a, b, acc[t]);
    }
  }

  const int nl = lane & 15;
  const int mb = row0 + ((lane & 16) ? 8 : 0);
#pragma unroll
  for (int t = 0; t < 8; ++t) {
    const int   n  = c0 + t * 16 + nl;
    const float bb = bo[n];
#pragma unroll
    for (int r = 0; r < 8; ++r) {
      out[(size_t)(mb + r) * H_DIM + n] = acc[t][r] + bb;
    }
  }
}

// ---------------------------------------------------------------------------
extern "C" void kernel_launch(void* const* d_in, const int* in_sizes, int n_in,
                              void* d_out, int out_size, void* d_ws,
                              size_t ws_size, hipStream_t stream) {
  const float* query = (const float*)d_in[0];
  // d_in[1] = attention_mask: uniform per-row shift -> softmax no-op.
  const float* Wq = (const float*)d_in[2];
  const float* bq = (const float*)d_in[3];
  const float* Wk = (const float*)d_in[4];
  const float* bk = (const float*)d_in[5];
  const float* Wv = (const float*)d_in[6];
  const float* bv = (const float*)d_in[7];
  const float* Wo = (const float*)d_in[8];
  const float* bo = (const float*)d_in[9];
  float*       out = (float*)d_out;

  // Workspace layout (f16): Qh | Kh | Vh | Wt | Woh
  _Float16* Qh  = (_Float16*)d_ws;
  _Float16* Kh  = Qh + (size_t)BL * P_DIM;
  _Float16* Vh  = Kh + (size_t)BL * P_DIM;
  _Float16* Wt  = Vh + (size_t)BL * P_DIM;
  _Float16* Woh = Wt + (size_t)BL * P_DIM;

  cvt_kernel<<<(P_DIM * H_DIM + 255) / 256, 256, 0, stream>>>(Wo, Woh,
                                                              P_DIM * H_DIM);
  qkv_kernel<<<BL / 32, 192, 0, stream>>>(query, Wq, bq, Wk, bk, Wv, bv, Qh,
                                          Kh, Vh);
  attn_kernel<<<dim3(L_DIM / 64, B_DIM), 128, 0, stream>>>(Qh, Kh, Vh, Wt);
  out_kernel<<<BL / 32, 256, 0, stream>>>(Wt, Woh, bo, out);
}
